// GINAttNet_51556787422033
// MI455X (gfx1250) — matrix operand order repaired and checked by reading, not compiled, not run on hardware
//
#include <hip/hip_runtime.h>
#include <math.h>
#include <stdint.h>

// ---------------------------------------------------------------------------
// GINAttNet-style GraphConv+TopK network, CDNA5 (gfx1250) implementation.
//  - Edge aggregation: wave-per-edge coalesced gather + f32 global atomics
//    (memory/atomic bound stage; ~0.6GB gathered reads at 23.3 TB/s HBM)
//  - All 4 GEMMs via V_WMMA_F32_16X16X32_F16 (f16 in, f32 accum), wave32.
//    Each wave holds ALL N-tile accumulators (A streamed once, 7x less HBM
//    traffic), weights staged to LDS once per workgroup via the Tensor Data
//    Mover (tensor_load_to_lds + s_wait_tensorcnt), B read via ds_load_b128.
// ---------------------------------------------------------------------------

typedef __attribute__((ext_vector_type(16))) _Float16 v16h;
typedef __attribute__((ext_vector_type(8)))  _Float16 v8h;
typedef __attribute__((ext_vector_type(8)))  float    v8f;
typedef __attribute__((ext_vector_type(4)))  uint32_t v4u;
typedef __attribute__((ext_vector_type(8)))  int      v8i;
typedef __attribute__((ext_vector_type(4)))  int      v4i;

#define B_GRAPHS 512
#define N_NODES  95
#define F_DIM    200
#define EPG      1520
#define E_EDGES  (B_GRAPHS * EPG)      // 778240
#define DIM2     100
#define K1       48
#define K2       24
#define DIM1     64
#define DIM4     64
#define EPS_BN   1e-5f

#define FP   224                       // F padded to mult of 32
#define D2P  128                       // DIM2 padded to mult of 32
#define NP1  112                       // DIM2 out-cols padded to mult of 16
#define M1   (B_GRAPHS * N_NODES)      // 48640  (mult of 16)
#define M2   (B_GRAPHS * K1)           // 24576
#define M3   (B_GRAPHS * K2)           // 12288

// d_out layout (flat, return order)
#define O_OUT1 0
#define O_OUT2 512
#define O_OUT3 1024
#define O_P1W  1536
#define O_P2W  1636
#define O_S1   1736
#define O_S2   26312
#define O_PM1  38600
#define O_PM2  63176

__device__ __forceinline__ float sigmoidf_(float x) { return 1.0f / (1.0f + __expf(-x)); }

// ------------------------------- utility -----------------------------------
__global__ void zero32_kernel(uint32_t* p, size_t n) {
  size_t i = (size_t)blockIdx.x * blockDim.x + threadIdx.x;
  if (i < n) p[i] = 0u;
}

// f32 [rows,cols] -> f16 [rows,colsP] (pad region must be pre-zeroed)
__global__ void cvt_pad_f16_kernel(const float* __restrict__ in, _Float16* __restrict__ out,
                                   int rows, int cols, int colsP) {
  size_t i = (size_t)blockIdx.x * blockDim.x + threadIdx.x;
  size_t n = (size_t)rows * cols;
  if (i >= n) return;
  int r = (int)(i / cols), c = (int)(i % cols);
  out[(size_t)r * colsP + c] = (_Float16)in[i];
}

// --------------------------- edge aggregation ------------------------------
// one wave per edge; lanes stride the feature dim (coalesced), f32 atomics.
__global__ void edge_agg1_kernel(const float* __restrict__ x,
                                 const int* __restrict__ src,
                                 const int* __restrict__ dst,
                                 float* __restrict__ agg) {
  int gid  = blockIdx.x * blockDim.x + threadIdx.x;
  int e    = gid >> 5;
  int lane = gid & 31;
  if (e >= E_EDGES) return;
  int s = src[e], d = dst[e];
  const float* xs = x + (size_t)s * F_DIM;
  float*       ad = agg + (size_t)d * F_DIM;
  __builtin_prefetch(xs, 0, 1);          // global_prefetch_b8
  for (int f = lane; f < F_DIM; f += 32) atomicAdd(&ad[f], xs[f]);
}

// stage-2 aggregation with relabel + validity mask
__global__ void edge_agg2_kernel(const float* __restrict__ h,
                                 const int* __restrict__ src,
                                 const int* __restrict__ dst,
                                 const int* __restrict__ kept,
                                 const int* __restrict__ new_idx,
                                 float* __restrict__ agg) {
  int gid  = blockIdx.x * blockDim.x + threadIdx.x;
  int e    = gid >> 5;
  int lane = gid & 31;
  if (e >= E_EDGES) return;
  int s = src[e], d = dst[e];
  if (!(kept[s] & kept[d])) return;
  int s2 = new_idx[s], d2 = new_idx[d];
  const float* hs = h + (size_t)s2 * DIM2;
  float*       ad = agg + (size_t)d2 * DIM2;
  __builtin_prefetch(hs, 0, 1);
  for (int f = lane; f < DIM2; f += 32) atomicAdd(&ad[f], hs[f]);
}

// ------------------- TDM: stage weights into LDS ---------------------------
// 1-D copy descriptor (D#): data_size=8B, count=1 (valid), type=2 ("image").
// Group0: [count | lds_addr | global_addr[56:0] | type]
// Group1: [mask=0,data_size=3 | dim0 | dim1=1 | tile0=n8 | tile1=1 | stride0]
__device__ __forceinline__ void tdm_stage_lds(const void* gptr, uint32_t lds_off,
                                              uint32_t nbytes) {
  uint64_t ga = (uint64_t)(uintptr_t)gptr;
  uint32_t n8 = nbytes >> 3;                       // elements of 8 bytes
  v4u g0;
  g0[0] = 1u;                                      // count=1, user-mode, no gather
  g0[1] = lds_off;                                 // lds_addr (bytes)
  g0[2] = (uint32_t)ga;                            // global_addr[31:0]
  g0[3] = ((uint32_t)(ga >> 32) & 0x01FFFFFFu)     // global_addr[56:32]
        | (2u << 30);                              // type=2
  v8i g1;
  g1[0] = 0x00030000;                              // workgroup_mask=0, data_size=3
  g1[1] = (int)((n8 & 0xFFFFu) << 16);             // tensor_dim0[15:0]
  g1[2] = (int)((n8 >> 16) | (1u << 16));          // tensor_dim0[31:16], tensor_dim1=1
  g1[3] = (int)(n8 << 16);                         // tile_dim0 = n8 (< 65536)
  g1[4] = 1;                                       // tile_dim1 = 1
  g1[5] = (int)n8;                                 // tensor_dim0_stride[31:0]
  g1[6] = 0;
  g1[7] = 0;
  v4i z4 = {0, 0, 0, 0};
#if __has_builtin(__builtin_amdgcn_tensor_load_to_lds)
 #if defined(__clang_major__) && (__clang_major__ >= 23)
  v8i z8 = {0, 0, 0, 0, 0, 0, 0, 0};
  __builtin_amdgcn_tensor_load_to_lds(g0, g1, z4, z4, z8, 0);
 #else
  __builtin_amdgcn_tensor_load_to_lds(g0, g1, z4, z4, 0);
 #endif
 #if __has_builtin(__builtin_amdgcn_s_wait_tensorcnt)
  __builtin_amdgcn_s_wait_tensorcnt(0);
 #endif
#else
  (void)g0; (void)g1; (void)z4;
#endif
}

// ------------------------------ WMMA GEMM ----------------------------------
// OUT[M,Ncols] = relu( A0*W0^T (+ A1*W1^T) + bias ).
// W staged in LDS: [ (DUAL?2:1) * NT*16, Kp ] f16 row-major ([W0 | W1]).
// One wave per M-tile; all NT N-tile accumulators live in VGPRs so each A
// fragment is loaded from global exactly once per k-step (NT-fold reuse).
// Fragment layouts used here (A/B share contiguous k order, not the ISA table order):
//   A 16x32: lane<16 row M=lane holds K={kb..kb+15}, kb=k0
//            (lanes>=16: kb=k0+16).
//   B 32x16: lane holds 16 contiguous K of column N=lane&15 == one contiguous
//            slice of the [out,in] weight row (lanes>=16: K base +16).
//   C/D: VGPR r -> row r + 8*(lane>=16), col lane&15.
template <int NT, bool DUAL>
__global__ void wmma_gemm_tile_kernel(const _Float16* __restrict__ A0,
                                      const _Float16* __restrict__ A1,
                                      const _Float16* __restrict__ W,
                                      const float* __restrict__ bias,
                                      float* __restrict__ OUT,
                                      int M, int Kp, int Ncols) {
  extern __shared__ _Float16 smem[];
  const uint32_t wbytes = (uint32_t)((DUAL ? 2 : 1) * NT * 16) * (uint32_t)Kp * 2u;

#if __has_builtin(__builtin_amdgcn_tensor_load_to_lds)
  if (threadIdx.x == 0) {
    tdm_stage_lds(W, (uint32_t)(uintptr_t)(void*)smem, wbytes);   // TDM async DMA
  }
  __syncthreads();
#else
  for (uint32_t i = threadIdx.x; i < (wbytes >> 2); i += blockDim.x)
    ((uint32_t*)smem)[i] = ((const uint32_t*)W)[i];
  __syncthreads();
#endif

  const int wave = threadIdx.x >> 5;
  const int lane = threadIdx.x & 31;
  const int hs   = lane >> 4;
  const int lr   = lane & 15;
  const int m0   = (blockIdx.x * (blockDim.x >> 5) + wave) * 16;
  if (m0 >= M) return;

  v8f acc[NT] = {};
  const _Float16* a0row = A0 + (size_t)(m0 + lr) * Kp;
  const _Float16* a1row = DUAL ? (A1 + (size_t)(m0 + lr) * Kp) : a0row;

  for (int k0 = 0; k0 < Kp; k0 += 32) {
    const int kb = k0 + hs * 16;
    v8h alo = *(const v8h*)(a0row + kb);
    v8h ahi = *(const v8h*)(a0row + kb + 8);
    v16h a;
#pragma unroll
    for (int i = 0; i < 8; ++i) { a[i] = alo[i]; a[8 + i] = ahi[i]; }
    v16h a2;
    if (DUAL) {
      v8h clo = *(const v8h*)(a1row + kb);
      v8h chi = *(const v8h*)(a1row + kb + 8);
#pragma unroll
      for (int i = 0; i < 8; ++i) { a2[i] = clo[i]; a2[8 + i] = chi[i]; }
    }
#pragma unroll
    for (int n = 0; n < NT; ++n) {
      const _Float16* brow = smem + (size_t)(n * 16 + lr) * Kp + hs * 16 + k0;
      v16h b = *(const v16h*)brow;                       // ds_load_b128 x2
      acc[n] = __builtin_amdgcn_wmma_f32_16x16x32_f16(false, a, false, b,
                                                      (short)0, acc[n], false, false);
      if (DUAL) {
        const _Float16* brow2 = smem + (size_t)(NT * 16 + n * 16 + lr) * Kp + hs * 16 + k0;
        v16h b2 = *(const v16h*)brow2;
        acc[n] = __builtin_amdgcn_wmma_f32_16x16x32_f16(false, a2, false, b2,
                                                        (short)0, acc[n], false, false);
      }
    }
  }
#pragma unroll
  for (int n = 0; n < NT; ++n) {
    const int col = n * 16 + lr;
    if (col < Ncols) {
      const float bv = bias[col];
#pragma unroll
      for (int r = 0; r < 8; ++r) {
        int row = m0 + hs * 8 + r;
        float v = acc[n][r] + bv;
        OUT[(size_t)row * Ncols + col] = fmaxf(v, 0.0f);  // relu (all call sites)
      }
    }
  }
}

// ------------------------------ pooling ------------------------------------
__global__ void norms_kernel(const float* __restrict__ p1w, const float* __restrict__ p2w,
                             float* __restrict__ norms) {
  if (threadIdx.x == 0) {
    float s = 0.f; for (int i = 0; i < DIM2; ++i) s += p1w[i] * p1w[i];
    norms[0] = sqrtf(s);
  } else if (threadIdx.x == 1) {
    float s = 0.f; for (int i = 0; i < DIM2; ++i) s += p2w[i] * p2w[i];
    norms[1] = sqrtf(s);
  }
}

__global__ void score_kernel(const float* __restrict__ h, const float* __restrict__ w,
                             const float* __restrict__ norms, int which,
                             float* __restrict__ scores, int n) {
  int i = blockIdx.x * blockDim.x + threadIdx.x;
  if (i >= n) return;
  const float* hr = h + (size_t)i * DIM2;
  float d = 0.f;
  for (int j = 0; j < DIM2; ++j) d += hr[j] * w[j];
  scores[i] = sigmoidf_(d / norms[which]);
}

// one block per graph; strict '>' + ascending scan == jax.lax.top_k tie-break.
__global__ void topk_kernel(const float* __restrict__ scores, int n_per, int k,
                            int* __restrict__ perm, float* __restrict__ vals) {
  __shared__ float sc[128];
  __shared__ int used[128];
  int b = blockIdx.x;
  for (int i = threadIdx.x; i < n_per; i += blockDim.x) {
    sc[i] = scores[b * n_per + i];
    used[i] = 0;
  }
  __syncthreads();
  if (threadIdx.x != 0) return;
  for (int t = 0; t < k; ++t) {
    int best = 0; float bv = -1e30f;
    for (int i = 0; i < n_per; ++i)
      if (!used[i] && sc[i] > bv) { bv = sc[i]; best = i; }
    used[best] = 1;
    perm[b * k + t] = b * n_per + best;
    vals[b * k + t] = bv;
  }
}

__global__ void gather_scale_kernel(const float* __restrict__ h, const int* __restrict__ perm,
                                    const float* __restrict__ vals, float* __restrict__ out,
                                    int rows) {
  size_t i = (size_t)blockIdx.x * blockDim.x + threadIdx.x;
  size_t n = (size_t)rows * DIM2;
  if (i >= n) return;
  int r = (int)(i / DIM2), c = (int)(i % DIM2);
  out[i] = h[(size_t)perm[r] * DIM2 + c] * vals[r];
}

__global__ void mark_kept_kernel(const int* __restrict__ perm, int n,
                                 int* __restrict__ kept, int* __restrict__ new_idx) {
  int i = blockIdx.x * blockDim.x + threadIdx.x;
  if (i >= n) return;
  int p = perm[i];
  kept[p] = 1;
  new_idx[p] = i;
}

// [B,k,DIM2] -> [B, 2*DIM2] (max || mean)
__global__ void global_pools_kernel(const float* __restrict__ h, int k, float* __restrict__ out) {
  int i = blockIdx.x * blockDim.x + threadIdx.x;
  if (i >= B_GRAPHS * DIM2) return;
  int b = i / DIM2, f = i % DIM2;
  const float* hb = h + ((size_t)b * k) * DIM2 + f;
  float mx = -1e30f, sm = 0.f;
  for (int j = 0; j < k; ++j) {
    float v = hb[(size_t)j * DIM2];
    mx = fmaxf(mx, v);
    sm += v;
  }
  out[(size_t)b * (2 * DIM2) + f]        = mx;
  out[(size_t)b * (2 * DIM2) + DIM2 + f] = sm / (float)k;
}

__global__ void add_kernel(const float* __restrict__ a, const float* __restrict__ b,
                           float* __restrict__ o, int n) {
  int i = blockIdx.x * blockDim.x + threadIdx.x;
  if (i < n) o[i] = a[i] + b[i];
}

// ------------------------------ batch norm ---------------------------------
__global__ void bn_kernel(float* __restrict__ h, const float* __restrict__ g,
                          const float* __restrict__ b, int rows, int cols) {
  int c = blockIdx.x * blockDim.x + threadIdx.x;
  if (c >= cols) return;
  float m = 0.f;
  for (int r = 0; r < rows; ++r) m += h[(size_t)r * cols + c];
  m /= (float)rows;
  float v = 0.f;
  for (int r = 0; r < rows; ++r) { float d = h[(size_t)r * cols + c] - m; v += d * d; }
  v /= (float)rows;
  float s = rsqrtf(v + EPS_BN);
  for (int r = 0; r < rows; ++r)
    h[(size_t)r * cols + c] = g[c] * (h[(size_t)r * cols + c] - m) * s + b[c];
}

// ------------------------------- heads -------------------------------------
__global__ void heads_kernel(const float* __restrict__ gfeat,
                             const float* __restrict__ h1W1, const float* __restrict__ h1b1,
                             const float* __restrict__ h1W2, const float* __restrict__ h1b2,
                             const float* __restrict__ h2W1, const float* __restrict__ h2b1,
                             const float* __restrict__ h2W2, const float* __restrict__ h2b2,
                             const float* __restrict__ h3W1, const float* __restrict__ h3b1,
                             const float* __restrict__ h3W2, const float* __restrict__ h3b2,
                             float* __restrict__ out) {
  int b = blockIdx.x * blockDim.x + threadIdx.x;
  if (b >= B_GRAPHS) return;
  const float* x = gfeat + (size_t)b * DIM4;
  float o1 = h1b2[0];
  for (int j = 0; j < 16; ++j) {
    float t = h1b1[j];
    for (int i = 0; i < DIM4; ++i) t += x[i] * h1W1[j * DIM4 + i];
    o1 += fmaxf(t, 0.f) * h1W2[j];
  }
  out[O_OUT1 + b] = sigmoidf_(o1);
  float o2 = h2b2[0];
  for (int j = 0; j < 32; ++j) {
    float t = h2b1[j];
    for (int i = 0; i < DIM4; ++i) t += x[i] * h2W1[j * DIM4 + i];
    o2 += fmaxf(t, 0.f) * h2W2[j];
  }
  out[O_OUT2 + b] = o2;
  float o3 = h3b2[0];
  for (int j = 0; j < 32; ++j) {
    float t = h3b1[j];
    for (int i = 0; i < DIM4; ++i) t += x[i] * h3W1[j * DIM4 + i];
    o3 += fmaxf(t, 0.f) * h3W2[j];
  }
  out[O_OUT3 + b] = o3;
}

// ------------------------------ finalize -----------------------------------
__global__ void finalize_kernel(const float* __restrict__ p1w, const float* __restrict__ p2w,
                                const float* __restrict__ vals1, const float* __restrict__ vals2,
                                const int* __restrict__ perm1, const int* __restrict__ perm2,
                                float* __restrict__ out) {
  int i = blockIdx.x * blockDim.x + threadIdx.x;
  if (i < DIM2) { out[O_P1W + i] = p1w[i]; out[O_P2W + i] = p2w[i]; }
  if (i < M2)   { out[O_S1 + i] = sigmoidf_(vals1[i]); out[O_PM1 + i] = (float)perm1[i]; }
  if (i < M3)   { out[O_S2 + i] = sigmoidf_(vals2[i]); out[O_PM2 + i] = (float)perm2[i]; }
}

// ---------------------------------------------------------------------------
static inline int cdiv(long long a, long long b) { return (int)((a + b - 1) / b); }

extern "C" void kernel_launch(void* const* d_in, const int* in_sizes, int n_in,
                              void* d_out, int out_size, void* d_ws, size_t ws_size,
                              hipStream_t stream) {
  const float* x    = (const float*)d_in[0];
  const int*   eidx = (const int*)d_in[1];
  const float* c1_Wr = (const float*)d_in[2];
  const float* c1_Wn = (const float*)d_in[3];
  const float* c1_b  = (const float*)d_in[4];
  const float* c2_Wr = (const float*)d_in[5];
  const float* c2_Wn = (const float*)d_in[6];
  const float* c2_b  = (const float*)d_in[7];
  const float* p1_w  = (const float*)d_in[8];
  const float* p2_w  = (const float*)d_in[9];
  const float* lin1_W = (const float*)d_in[10];
  const float* lin1_b = (const float*)d_in[11];
  const float* bn1_g  = (const float*)d_in[12];
  const float* bn1_b  = (const float*)d_in[13];
  const float* lin2_W = (const float*)d_in[14];
  const float* lin2_b = (const float*)d_in[15];
  const float* bn2_g  = (const float*)d_in[16];
  const float* bn2_b  = (const float*)d_in[17];
  const float* h1W1 = (const float*)d_in[18]; const float* h1b1 = (const float*)d_in[19];
  const float* h1W2 = (const float*)d_in[20]; const float* h1b2 = (const float*)d_in[21];
  const float* h2W1 = (const float*)d_in[22]; const float* h2b1 = (const float*)d_in[23];
  const float* h2W2 = (const float*)d_in[24]; const float* h2b2 = (const float*)d_in[25];
  const float* h3W1 = (const float*)d_in[26]; const float* h3b1 = (const float*)d_in[27];
  const float* h3W2 = (const float*)d_in[28]; const float* h3b2 = (const float*)d_in[29];

  const int* src = eidx;
  const int* dst = eidx + E_EDGES;
  float* out = (float*)d_out;

  char* wsb = (char*)d_ws;
  size_t off = 0;
  auto alloc = [&](size_t bytes) -> void* {
    void* p = wsb + off;
    off = (off + bytes + 255) & ~(size_t)255;
    return p;
  };
  _Float16* xh    = (_Float16*)alloc((size_t)M1 * FP * 2);
  _Float16* aggh1 = (_Float16*)alloc((size_t)M1 * FP * 2);
  _Float16* wc1h  = (_Float16*)alloc((size_t)2 * NP1 * FP * 2);  // [Wr | Wn]
  float*    agg1  = (float*)alloc((size_t)M1 * F_DIM * 4);
  float*    h1    = (float*)alloc((size_t)M1 * DIM2 * 4);
  float*    sc1   = (float*)alloc((size_t)M1 * 4);
  int*      perm1 = (int*)alloc((size_t)M2 * 4);
  float*    vals1 = (float*)alloc((size_t)M2 * 4);
  float*    hp1   = (float*)alloc((size_t)M2 * DIM2 * 4);
  float*    x1    = (float*)alloc((size_t)B_GRAPHS * 2 * DIM2 * 4);
  int*      kept  = (int*)alloc((size_t)M1 * 4);
  int*      nidx  = (int*)alloc((size_t)M1 * 4);
  _Float16* hph   = (_Float16*)alloc((size_t)M2 * D2P * 2);
  _Float16* aggh2 = (_Float16*)alloc((size_t)M2 * D2P * 2);
  _Float16* wc2h  = (_Float16*)alloc((size_t)2 * NP1 * D2P * 2);
  float*    agg2  = (float*)alloc((size_t)M2 * DIM2 * 4);
  float*    h2    = (float*)alloc((size_t)M2 * DIM2 * 4);
  float*    sc2   = (float*)alloc((size_t)M2 * 4);
  int*      perm2 = (int*)alloc((size_t)M3 * 4);
  float*    vals2 = (float*)alloc((size_t)M3 * 4);
  float*    hp2   = (float*)alloc((size_t)M3 * DIM2 * 4);
  float*    x2    = (float*)alloc((size_t)B_GRAPHS * 2 * DIM2 * 4);
  float*    gsum  = (float*)alloc((size_t)B_GRAPHS * 2 * DIM2 * 4);
  _Float16* gh    = (_Float16*)alloc((size_t)B_GRAPHS * FP * 2);
  _Float16* wl1h  = (_Float16*)alloc((size_t)DIM1 * FP * 2);
  float*    g1    = (float*)alloc((size_t)B_GRAPHS * DIM1 * 4);
  _Float16* g1h   = (_Float16*)alloc((size_t)B_GRAPHS * DIM1 * 2);
  _Float16* wl2h  = (_Float16*)alloc((size_t)DIM4 * DIM1 * 2);
  float*    g2    = (float*)alloc((size_t)B_GRAPHS * DIM4 * 4);
  float*    norms = (float*)alloc(256);
  (void)ws_size; (void)in_sizes; (void)n_in; (void)out_size;

  const int TB = 256;
  auto zero = [&](void* p, size_t bytes) {
    size_t n32 = bytes / 4;
    zero32_kernel<<<cdiv((long long)n32, TB), TB, 0, stream>>>((uint32_t*)p, n32);
  };

  // ===== stage 1: GraphConv1 =====
  zero(xh,    (size_t)M1 * FP * 2);
  zero(aggh1, (size_t)M1 * FP * 2);
  zero(wc1h,  (size_t)2 * NP1 * FP * 2);
  zero(agg1,  (size_t)M1 * F_DIM * 4);
  cvt_pad_f16_kernel<<<cdiv((long long)M1 * F_DIM, TB), TB, 0, stream>>>(x, xh, M1, F_DIM, FP);
  cvt_pad_f16_kernel<<<cdiv((long long)DIM2 * F_DIM, TB), TB, 0, stream>>>(c1_Wr, wc1h, DIM2, F_DIM, FP);
  cvt_pad_f16_kernel<<<cdiv((long long)DIM2 * F_DIM, TB), TB, 0, stream>>>(c1_Wn, wc1h + (size_t)NP1 * FP, DIM2, F_DIM, FP);
  edge_agg1_kernel<<<cdiv((long long)E_EDGES * 32, TB), TB, 0, stream>>>(x, src, dst, agg1);
  cvt_pad_f16_kernel<<<cdiv((long long)M1 * F_DIM, TB), TB, 0, stream>>>(agg1, aggh1, M1, F_DIM, FP);
  {
    // A0=xh pairs W[0]=Wr ; A1=aggh1 pairs W[1]=Wn
    size_t smemB = (size_t)2 * NP1 * FP * 2;                  // 100,352 B
    wmma_gemm_tile_kernel<7, true><<<M1 / 16 / 8, 256, smemB, stream>>>(
        xh, aggh1, wc1h, c1_b, h1, M1, FP, DIM2);
  }

  // ===== pool1 =====
  norms_kernel<<<1, 32, 0, stream>>>(p1_w, p2_w, norms);
  score_kernel<<<cdiv(M1, TB), TB, 0, stream>>>(h1, p1_w, norms, 0, sc1, M1);
  topk_kernel<<<B_GRAPHS, 32, 0, stream>>>(sc1, N_NODES, K1, perm1, vals1);
  gather_scale_kernel<<<cdiv((long long)M2 * DIM2, TB), TB, 0, stream>>>(h1, perm1, vals1, hp1, M2);
  global_pools_kernel<<<cdiv(B_GRAPHS * DIM2, TB), TB, 0, stream>>>(hp1, K1, x1);

  // ===== relabel + GraphConv2 =====
  zero(kept, (size_t)M1 * 4);
  zero(nidx, (size_t)M1 * 4);
  mark_kept_kernel<<<cdiv(M2, TB), TB, 0, stream>>>(perm1, M2, kept, nidx);
  zero(hph,   (size_t)M2 * D2P * 2);
  zero(aggh2, (size_t)M2 * D2P * 2);
  zero(wc2h,  (size_t)2 * NP1 * D2P * 2);
  zero(agg2,  (size_t)M2 * DIM2 * 4);
  cvt_pad_f16_kernel<<<cdiv((long long)M2 * DIM2, TB), TB, 0, stream>>>(hp1, hph, M2, DIM2, D2P);
  cvt_pad_f16_kernel<<<cdiv((long long)DIM2 * DIM2, TB), TB, 0, stream>>>(c2_Wr, wc2h, DIM2, DIM2, D2P);
  cvt_pad_f16_kernel<<<cdiv((long long)DIM2 * DIM2, TB), TB, 0, stream>>>(c2_Wn, wc2h + (size_t)NP1 * D2P, DIM2, DIM2, D2P);
  edge_agg2_kernel<<<cdiv((long long)E_EDGES * 32, TB), TB, 0, stream>>>(hp1, src, dst, kept, nidx, agg2);
  cvt_pad_f16_kernel<<<cdiv((long long)M2 * DIM2, TB), TB, 0, stream>>>(agg2, aggh2, M2, DIM2, D2P);
  {
    size_t smemB = (size_t)2 * NP1 * D2P * 2;                 // 57,344 B
    wmma_gemm_tile_kernel<7, true><<<M2 / 16 / 8, 256, smemB, stream>>>(
        hph, aggh2, wc2h, c2_b, h2, M2, D2P, DIM2);
  }

  // ===== pool2 =====
  score_kernel<<<cdiv(M2, TB), TB, 0, stream>>>(h2, p2_w, norms, 1, sc2, M2);
  topk_kernel<<<B_GRAPHS, 32, 0, stream>>>(sc2, K1, K2, perm2, vals2);
  gather_scale_kernel<<<cdiv((long long)M3 * DIM2, TB), TB, 0, stream>>>(h2, perm2, vals2, hp2, M3);
  global_pools_kernel<<<cdiv(B_GRAPHS * DIM2, TB), TB, 0, stream>>>(hp2, K2, x2);

  // ===== MLP trunk =====
  add_kernel<<<cdiv(B_GRAPHS * 2 * DIM2, TB), TB, 0, stream>>>(x1, x2, gsum, B_GRAPHS * 2 * DIM2);
  zero(gh,   (size_t)B_GRAPHS * FP * 2);
  zero(wl1h, (size_t)DIM1 * FP * 2);
  cvt_pad_f16_kernel<<<cdiv((long long)B_GRAPHS * 2 * DIM2, TB), TB, 0, stream>>>(gsum, gh, B_GRAPHS, 2 * DIM2, FP);
  cvt_pad_f16_kernel<<<cdiv((long long)DIM1 * 2 * DIM2, TB), TB, 0, stream>>>(lin1_W, wl1h, DIM1, 2 * DIM2, FP);
  {
    size_t smemB = (size_t)DIM1 * FP * 2;                     // 28,672 B
    wmma_gemm_tile_kernel<4, false><<<B_GRAPHS / 16 / 8, 256, smemB, stream>>>(
        gh, gh, wl1h, lin1_b, g1, B_GRAPHS, FP, DIM1);
  }
  bn_kernel<<<cdiv(DIM1, 64), 64, 0, stream>>>(g1, bn1_g, bn1_b, B_GRAPHS, DIM1);
  cvt_pad_f16_kernel<<<cdiv((long long)B_GRAPHS * DIM1, TB), TB, 0, stream>>>(g1, g1h, B_GRAPHS, DIM1, DIM1);
  cvt_pad_f16_kernel<<<cdiv((long long)DIM4 * DIM1, TB), TB, 0, stream>>>(lin2_W, wl2h, DIM4, DIM1, DIM1);
  {
    size_t smemB = (size_t)DIM4 * DIM1 * 2;                   // 8,192 B
    wmma_gemm_tile_kernel<4, false><<<B_GRAPHS / 16 / 8, 256, smemB, stream>>>(
        g1h, g1h, wl2h, lin2_b, g2, B_GRAPHS, DIM1, DIM4);
  }
  bn_kernel<<<cdiv(DIM4, 64), 64, 0, stream>>>(g2, bn2_g, bn2_b, B_GRAPHS, DIM4);

  // ===== heads + outputs =====
  heads_kernel<<<cdiv(B_GRAPHS, TB), TB, 0, stream>>>(g2,
      h1W1, h1b1, h1W2, h1b2, h2W1, h2b1, h2W2, h2b2, h3W1, h3b1, h3W2, h3b2, out);
  finalize_kernel<<<cdiv(M2, TB), TB, 0, stream>>>(p1_w, p2_w, vals1, vals2, perm1, perm2, out);
}
